// Block_24644522344922
// MI455X (gfx1250) — matrix operand order repaired and checked by reading, not compiled, not run on hardware
//
#include <hip/hip_runtime.h>
#include <hip/hip_bf16.h>
#include <math.h>

// ---------------- types ----------------
typedef __bf16 bf16_t;
typedef __attribute__((ext_vector_type(8)))  __bf16 v8bf;
typedef __attribute__((ext_vector_type(16))) __bf16 v16bf;
typedef __attribute__((ext_vector_type(8)))  float  v8f;
typedef __attribute__((ext_vector_type(4)))  unsigned int u32x4;
typedef __attribute__((ext_vector_type(8)))  int i32x8;
typedef __attribute__((ext_vector_type(4)))  int i32x4;

// ---------------- problem constants ----------------
#define BIMG   8
#define HW     112
#define NTOK   12544          // 112*112
#define CCH    256
#define NHEAD  8
#define HD     32
#define WSZ    7
#define WSWS   49
#define WTOT   2048           // BIMG * 256 windows
#define MROWS  100352         // WTOT*49 == BIMG*NTOK
#define MSLACK 100416         // MROWS + 64 (load slack)
#define QKSCALE 0.17677669529663687f  // 1/sqrt(32)

// ---------------- helpers ----------------
static __device__ inline v8f vzero8() {
  v8f z = {0.f,0.f,0.f,0.f,0.f,0.f,0.f,0.f};
  return z;
}

static __device__ inline float wred32(float v) {
  for (int m = 16; m > 0; m >>= 1) v += __shfl_xor(v, m, 32);
  return v;
}

// A fragment for v_wmma_f32_16x16x32_bf16 from row-major A (lda elements/row).
static __device__ inline v16bf ld_afrag(const bf16_t* A, int lda, int m0, int kk, int lane) {
  int hi = lane >> 4;
  const bf16_t* p = A + (size_t)(m0 + (lane & 15)) * lda + kk + hi * 8;
  v8bf lo = *(const v8bf*)p;
  v8bf hb = *(const v8bf*)(p + 16);
  return __builtin_shufflevector(lo, hb, 0,1,2,3,4,5,6,7,8,9,10,11,12,13,14,15);
}

// B fragment from BT (N-major, ldb elements/row = K).
static __device__ inline v16bf ld_bfrag(const bf16_t* BT, int ldb, int n0, int kk, int lane) {
  int hi = lane >> 4;
  const bf16_t* p = BT + (size_t)(n0 + (lane & 15)) * ldb + kk + hi * 8;
  v8bf lo = *(const v8bf*)p;
  v8bf hb = *(const v8bf*)(p + 16);
  return __builtin_shufflevector(lo, hb, 0,1,2,3,4,5,6,7,8,9,10,11,12,13,14,15);
}

static __device__ inline v8f wmma_bf16(v16bf a, v16bf b, v8f c) {
  return __builtin_amdgcn_wmma_f32_16x16x32_bf16(false, a, false, b, (short)0, c, false, false);
}

// window row -> image row (roll(-3,-3) + 7x7 partition; same map used for inverse scatter)
static __device__ inline int win_row_to_img_row(int m) {
  int w = m / WSWS, tok = m % WSWS;
  int bimg = w >> 8, wi = w & 255;
  int wh = wi >> 4, ww = wi & 15;
  int th = tok / WSZ, tw = tok % WSZ;
  int hs = (wh * WSZ + th + 3) % HW;
  int ws = (ww * WSZ + tw + 3) % HW;
  return bimg * NTOK + hs * HW + ws;
}

// ---- Tensor Data Mover: 1-D contiguous tile (nelem bf16) global -> LDS ----
// amdgpu-toolchain (clang-23) 6-arg form:
//   (uint32x4 g0, int32x8 g1, int32x4 g2, int32x4 g3, int32x8 extra, i32 cpol)
#if defined(__gfx1250__) && __has_builtin(__builtin_amdgcn_tensor_load_to_lds)
#define USE_TDM 1
static __device__ inline void tdm_load_1d(const bf16_t* gsrc, void* lds_dst, int nelem) {
  unsigned lds = (unsigned)(size_t)lds_dst;            // low 32 bits = LDS offset
  unsigned long long ga = (unsigned long long)(size_t)gsrc;
  // D# group0: count=1 | lds_addr | global_addr[56:0] | type=2
  u32x4 g0 = { 1u,
               lds,
               (unsigned)(ga & 0xffffffffull),
               ((unsigned)((ga >> 32) & 0x01ffffffull)) | 0x80000000u };
  // D# group1: data_size=1(2B); tensor_dim0=nelem; tensor_dim1=1; tile_dim0=nelem;
  //            tile_dim1=1; tensor_dim0_stride=nelem
  i32x8 g1 = { (int)(1u << 16),
               (int)((nelem & 0xffff) << 16),
               (int)(((nelem >> 16) & 0xffff) | (1 << 16)),
               (int)((nelem & 0xffff) << 16),
               1,
               nelem,
               0, 0 };
  i32x4 z4 = {0, 0, 0, 0};
  i32x8 z8 = {0, 0, 0, 0, 0, 0, 0, 0};
  __builtin_amdgcn_tensor_load_to_lds(g0, g1, z4, z4, z8, 0);
}
#else
#define USE_TDM 0
#endif

// ---------------- weight convert: W[K][N] f32 -> BT[N][K] bf16 ----------------
__global__ __launch_bounds__(256) void k_convT(const float* __restrict__ W,
                                               bf16_t* __restrict__ BT, int K, int N) {
  int idx = blockIdx.x * 256 + threadIdx.x;
  if (idx < K * N) {
    int n = idx / K, k = idx - n * K;
    BT[idx] = (bf16_t)W[(size_t)k * N + n];
  }
}

// ---------------- LayerNorm (optionally gathering windowed/rolled rows) ----------------
template<bool GATHER>
__global__ __launch_bounds__(256) void k_ln(const float* __restrict__ x,
                                            const float* __restrict__ g,
                                            const float* __restrict__ b,
                                            bf16_t* __restrict__ out) {
  int wv = threadIdx.x >> 5, lane = threadIdx.x & 31;
  int m = blockIdx.x * 8 + wv;
  if (m >= MROWS) return;
  int src = GATHER ? win_row_to_img_row(m) : m;
  const float4* p = (const float4*)(x + (size_t)src * CCH) + lane * 2;
  float4 v0 = p[0], v1 = p[1];
  float vals[8] = {v0.x, v0.y, v0.z, v0.w, v1.x, v1.y, v1.z, v1.w};
  float s = 0.f, s2 = 0.f;
#pragma unroll
  for (int i = 0; i < 8; ++i) { s += vals[i]; s2 += vals[i] * vals[i]; }
  s = wred32(s); s2 = wred32(s2);
  float mean = s * (1.f / CCH);
  float var  = s2 * (1.f / CCH) - mean * mean;
  float inv  = 1.f / sqrtf(var + 1e-5f);
  bf16_t* o = out + (size_t)m * CCH + lane * 8;
#pragma unroll
  for (int i = 0; i < 8; ++i) {
    int c = lane * 8 + i;
    o[i] = (bf16_t)((vals[i] - mean) * inv * g[c] + b[c]);
  }
}

// ---------------- QKV GEMM (M x 768 x 256) with q/k/vT scatter epilogue ----------------
// block tile 64x128, 4 waves, wave tile 32x64 (2 M-frags x 4 N-frags)
__global__ __launch_bounds__(128) void k_gemm_qkv(const bf16_t* __restrict__ A,
                                                  const bf16_t* __restrict__ BT,
                                                  const float* __restrict__ bias,
                                                  bf16_t* __restrict__ Qb,
                                                  bf16_t* __restrict__ Kb,
                                                  bf16_t* __restrict__ VTb) {
  const int K = CCH;
  int lane = threadIdx.x & 31, wv = threadIdx.x >> 5;
  int mb = blockIdx.y * 64 + 32 * (wv >> 1);
  int nb = blockIdx.x * 128 + 64 * (wv & 1);
  v8f acc[2][4];
#pragma unroll
  for (int i = 0; i < 2; ++i)
#pragma unroll
    for (int j = 0; j < 4; ++j) acc[i][j] = vzero8();

#pragma unroll 4
  for (int kk = 0; kk < K; kk += 32) {
    // branch-free speculative prefetch of next K-slab (OOB prefetch is dropped by HW)
    __builtin_prefetch(A  + (size_t)(mb + (lane & 15)) * K + kk + 32, 0, 1);
    __builtin_prefetch(BT + (size_t)(nb + (lane & 15)) * K + kk + 32, 0, 1);
    v16bf a0 = ld_afrag(A, K, mb,      kk, lane);
    v16bf a1 = ld_afrag(A, K, mb + 16, kk, lane);
    v16bf bf[4];
#pragma unroll
    for (int nt = 0; nt < 4; ++nt) bf[nt] = ld_bfrag(BT, K, nb + 16 * nt, kk, lane);
#pragma unroll
    for (int nt = 0; nt < 4; ++nt) {
      acc[0][nt] = wmma_bf16(a0, bf[nt], acc[0][nt]);
      acc[1][nt] = wmma_bf16(a1, bf[nt], acc[1][nt]);
    }
  }
  int hi = lane >> 4, ln = lane & 15;
#pragma unroll
  for (int mi = 0; mi < 2; ++mi)
#pragma unroll
    for (int nt = 0; nt < 4; ++nt)
#pragma unroll
      for (int j = 0; j < 8; ++j) {
        int row = mb + 16 * mi + 8 * hi + j;
        if (row >= MROWS) continue;
        int col = nb + 16 * nt + ln;
        float v = acc[mi][nt][j] + bias[col];
        int which = col >> 8, head = (col >> 5) & 7, d = col & 31;
        int w = row / WSWS, tok = row - w * WSWS;
        size_t base = (size_t)(w * NHEAD + head) * 2048;   // 64x32 tile
        if (which == 0)      Qb[base + tok * HD + d] = (bf16_t)(v * QKSCALE);
        else if (which == 1) Kb[base + tok * HD + d] = (bf16_t)v;
        else                 VTb[base + d * 64 + tok] = (bf16_t)v; // transposed, tok padded to 64
      }
}

// ---------------- fused windowed attention: one block per (window, head) ----------------
__global__ __launch_bounds__(128) void k_attn(const bf16_t* __restrict__ Q,
                                              const bf16_t* __restrict__ Kb,
                                              const bf16_t* __restrict__ VTb,
                                              const float* __restrict__ mask,
                                              bf16_t* __restrict__ out) {
  __shared__ __align__(32) float  sS[4][16][64];
  __shared__ __align__(32) bf16_t sP[4][16][64];
#if USE_TDM
  __shared__ __align__(32) bf16_t sQ[2048];
  __shared__ __align__(32) bf16_t sK[2048];
  __shared__ __align__(32) bf16_t sV[2048];
#endif
  int lane = threadIdx.x & 31, mt = threadIdx.x >> 5;
  int wh = blockIdx.x;
  int w = wh >> 3, h = wh & 7;
  const bf16_t* qg  = Q   + (size_t)wh * 2048;
  const bf16_t* kg  = Kb  + (size_t)wh * 2048;
  const bf16_t* vg  = VTb + (size_t)wh * 2048;
  int hi = lane >> 4, ln = lane & 15;

#if USE_TDM
  // TDM: DMA the three contiguous 4KB tiles into LDS, wave 0 owns the descriptors.
  if (threadIdx.x < 32) {
    tdm_load_1d(qg, sQ, 2048);
    tdm_load_1d(kg, sK, 2048);
    tdm_load_1d(vg, sV, 2048);
    __builtin_amdgcn_s_wait_tensorcnt(0);
  }
  __syncthreads();
  const bf16_t* qp = sQ;
  const bf16_t* kp = sK;
  const bf16_t* vp = sV;
#else
  const bf16_t* qp = qg;
  const bf16_t* kp = kg;
  const bf16_t* vp = vg;
#endif

  // S = q @ k^T  (hd=32 contraction: one WMMA K-step)
  v16bf aq = ld_afrag(qp, HD, 16 * mt, 0, lane);
#pragma unroll
  for (int nt = 0; nt < 4; ++nt) {
    v16bf bk = ld_bfrag(kp, HD, 16 * nt, 0, lane);
    v8f c = vzero8();
    c = wmma_bf16(aq, bk, c);
#pragma unroll
    for (int j = 0; j < 8; ++j)
      sS[mt][8 * hi + j][16 * nt + ln] = c[j];
  }
  __syncthreads();

  // masked softmax over 49 valid columns; pack bf16 probs (cols>=49 -> 0)
  if (lane < 16) {
    int r = lane, rq = 16 * mt + r;
    if (rq < WSWS) {
      int wi = w & 255;
      const float* mrow = mask + (size_t)wi * (WSWS * WSWS) + rq * WSWS;
      float mx = -1e30f;
      for (int c = 0; c < WSWS; ++c) {
        float t = sS[mt][r][c] + mrow[c];
        sS[mt][r][c] = t;
        mx = fmaxf(mx, t);
      }
      float sum = 0.f;
      for (int c = 0; c < WSWS; ++c) {
        float e = __expf(sS[mt][r][c] - mx);
        sS[mt][r][c] = e;
        sum += e;
      }
      float inv = 1.f / sum;
      for (int c = 0; c < WSWS; ++c) sP[mt][r][c] = (bf16_t)(sS[mt][r][c] * inv);
      for (int c = WSWS; c < 64; ++c) sP[mt][r][c] = (bf16_t)0.f;
    } else {
      for (int c = 0; c < 64; ++c) sP[mt][r][c] = (bf16_t)0.f;
    }
  }
  __syncthreads();

  // O = P @ V  (K=64 padded: two WMMA K-steps; N=32: two N-tiles)
#pragma unroll
  for (int nt2 = 0; nt2 < 2; ++nt2) {
    v8f acc = vzero8();
#pragma unroll
    for (int ks = 0; ks < 2; ++ks) {
      int kk2 = ks * 32;
      const bf16_t* pp = &sP[mt][0][0] + ln * 64 + kk2 + 8 * hi;
      v8bf lo = *(const v8bf*)pp;
      v8bf hb = *(const v8bf*)(pp + 16);
      v16bf pa = __builtin_shufflevector(lo, hb, 0,1,2,3,4,5,6,7,8,9,10,11,12,13,14,15);
      v16bf bv = ld_bfrag(vp, 64, 16 * nt2, kk2, lane);
      acc = wmma_bf16(pa, bv, acc);
    }
#pragma unroll
    for (int j = 0; j < 8; ++j) {
      int tok = 16 * mt + 8 * hi + j;
      if (tok < WSWS)
        out[((size_t)w * WSWS + tok) * CCH + h * HD + 16 * nt2 + ln] = (bf16_t)acc[j];
    }
  }
}

// ---------------- generic GEMM with epilogues ----------------
// block tile 64x128, wave tile 32x64.
// EPI 1: GELU(exact) -> bf16 ; EPI 2: +res -> f32 (same index) ; EPI 3: proj scatter +res -> f32
template<int EPI>
__global__ __launch_bounds__(128) void k_gemm(const bf16_t* __restrict__ A,
                                              const bf16_t* __restrict__ BT,
                                              const float* __restrict__ bias,
                                              void* __restrict__ outp,
                                              const float* __restrict__ res,
                                              int M, int N, int K) {
  int lane = threadIdx.x & 31, wv = threadIdx.x >> 5;
  int mb = blockIdx.y * 64 + 32 * (wv >> 1);
  int nb = blockIdx.x * 128 + 64 * (wv & 1);
  v8f acc[2][4];
#pragma unroll
  for (int i = 0; i < 2; ++i)
#pragma unroll
    for (int j = 0; j < 4; ++j) acc[i][j] = vzero8();

#pragma unroll 4
  for (int kk = 0; kk < K; kk += 32) {
    // branch-free speculative prefetch of next K-slab (OOB prefetch is dropped by HW)
    __builtin_prefetch(A  + (size_t)(mb + (lane & 15)) * K + kk + 32, 0, 1);
    __builtin_prefetch(BT + (size_t)(nb + (lane & 15)) * K + kk + 32, 0, 1);
    v16bf a0 = ld_afrag(A, K, mb,      kk, lane);
    v16bf a1 = ld_afrag(A, K, mb + 16, kk, lane);
    v16bf bf[4];
#pragma unroll
    for (int nt = 0; nt < 4; ++nt) bf[nt] = ld_bfrag(BT, K, nb + 16 * nt, kk, lane);
#pragma unroll
    for (int nt = 0; nt < 4; ++nt) {
      acc[0][nt] = wmma_bf16(a0, bf[nt], acc[0][nt]);
      acc[1][nt] = wmma_bf16(a1, bf[nt], acc[1][nt]);
    }
  }
  int hi = lane >> 4, ln = lane & 15;
#pragma unroll
  for (int mi = 0; mi < 2; ++mi)
#pragma unroll
    for (int nt = 0; nt < 4; ++nt)
#pragma unroll
      for (int j = 0; j < 8; ++j) {
        int row = mb + 16 * mi + 8 * hi + j;
        if (row >= M) continue;
        int col = nb + 16 * nt + ln;
        float v = acc[mi][nt][j] + bias[col];
        if (EPI == 1) {
          float ge = 0.5f * v * (1.0f + erff(v * 0.70710678118654752f));
          ((bf16_t*)outp)[(size_t)row * N + col] = (bf16_t)ge;
        } else if (EPI == 2) {
          size_t idx = (size_t)row * N + col;
          ((float*)outp)[idx] = res[idx] + v;
        } else { // EPI == 3: window-reverse + roll(+3,+3) scatter with residual
          size_t idx = (size_t)win_row_to_img_row(row) * CCH + col;
          ((float*)outp)[idx] = res[idx] + v;
        }
      }
}

// ---------------- host launcher ----------------
extern "C" void kernel_launch(void* const* d_in, const int* in_sizes, int n_in,
                              void* d_out, int out_size, void* d_ws, size_t ws_size,
                              hipStream_t stream) {
  const float* x      = (const float*)d_in[0];
  const float* mask   = (const float*)d_in[1];
  const float* n1g    = (const float*)d_in[2];
  const float* n1b    = (const float*)d_in[3];
  const float* qkv_w  = (const float*)d_in[4];
  const float* qkv_b  = (const float*)d_in[5];
  const float* proj_w = (const float*)d_in[6];
  const float* proj_b = (const float*)d_in[7];
  const float* n2g    = (const float*)d_in[8];
  const float* n2b    = (const float*)d_in[9];
  const float* fc1_w  = (const float*)d_in[10];
  const float* fc1_b  = (const float*)d_in[11];
  const float* fc2_w  = (const float*)d_in[12];
  const float* fc2_b  = (const float*)d_in[13];

  char* ws = (char*)d_ws;
  size_t off = 0;
  auto alloc = [&](size_t bytes) -> char* {
    char* p = ws + off;
    off += (bytes + 255) & ~(size_t)255;
    return p;
  };

  bf16_t* qkvT  = (bf16_t*)alloc((size_t)768 * 256 * 2);
  bf16_t* projT = (bf16_t*)alloc((size_t)256 * 256 * 2);
  bf16_t* fc1T  = (bf16_t*)alloc((size_t)1024 * 256 * 2);
  bf16_t* fc2T  = (bf16_t*)alloc((size_t)256 * 1024 * 2);
  bf16_t* buf1  = (bf16_t*)alloc((size_t)MSLACK * 256 * 2);   // xw, then attnout
  float*  xres  = (float*) alloc((size_t)MROWS * 256 * 4);
  bf16_t* h2    = (bf16_t*)alloc((size_t)MSLACK * 256 * 2);
  // big region: Q/K/VT during attention, reused as FC1 hidden afterwards
  size_t qkv_bytes = (size_t)3 * WTOT * NHEAD * 2048 * 2;     // 3 x 64 MiB
  size_t hid_bytes = (size_t)MSLACK * 1024 * 2;
  char*  Rbig  = alloc(qkv_bytes > hid_bytes ? qkv_bytes : hid_bytes);
  bf16_t* Qb   = (bf16_t*)Rbig;
  bf16_t* Kb   = (bf16_t*)(Rbig + (size_t)WTOT * NHEAD * 2048 * 2);
  bf16_t* VTb  = (bf16_t*)(Rbig + (size_t)2 * WTOT * NHEAD * 2048 * 2);
  bf16_t* hid  = (bf16_t*)Rbig;

  // 1) weights -> bf16 transposed
  k_convT<<<768, 256, 0, stream>>>(qkv_w, qkvT, 256, 768);
  k_convT<<<256, 256, 0, stream>>>(proj_w, projT, 256, 256);
  k_convT<<<1024, 256, 0, stream>>>(fc1_w, fc1T, 256, 1024);
  k_convT<<<1024, 256, 0, stream>>>(fc2_w, fc2T, 1024, 256);

  // 2) LN1 + roll + window partition -> bf16
  k_ln<true><<<MROWS / 8, 256, 0, stream>>>(x, n1g, n1b, buf1);

  // 3) zero VT padding region (tok 49..63 must be exactly 0)
  (void)hipMemsetAsync(VTb, 0, (size_t)WTOT * NHEAD * 2048 * 2, stream);

  // 4) QKV GEMM
  k_gemm_qkv<<<dim3(768 / 128, MSLACK / 64), 128, 0, stream>>>(buf1, qkvT, qkv_b, Qb, Kb, VTb);

  // 5) attention (one block per window-head)
  k_attn<<<WTOT * NHEAD, 128, 0, stream>>>(Qb, Kb, VTb, mask, buf1);

  // 6) proj + window reverse + unshift + residual
  k_gemm<3><<<dim3(256 / 128, MSLACK / 64), 128, 0, stream>>>(buf1, projT, proj_b, xres, x,
                                                              MROWS, 256, 256);

  // 7) LN2
  k_ln<false><<<MROWS / 8, 256, 0, stream>>>(xres, n2g, n2b, h2);

  // 8) FC1 + GELU
  k_gemm<1><<<dim3(1024 / 128, MSLACK / 64), 128, 0, stream>>>(h2, fc1T, fc1_b, hid, nullptr,
                                                               MROWS, 1024, 256);

  // 9) FC2 + residual -> f32 output
  k_gemm<2><<<dim3(256 / 128, MSLACK / 64), 128, 0, stream>>>(hid, fc2T, fc2_b, d_out, xres,
                                                              MROWS, 256, 1024);
}
